// CGCNN_Original_19456201850982
// MI455X (gfx1250) — compile-verified, matrix operand order checked by reading, not run on hardware
//
#include <hip/hip_runtime.h>
#include <hip/hip_bf16.h>

typedef __attribute__((ext_vector_type(16))) _Float16 v16h;
typedef __attribute__((ext_vector_type(8)))  _Float16 v8h;
typedef __attribute__((ext_vector_type(4)))  _Float16 v4h;
typedef __attribute__((ext_vector_type(8)))  float    v8f;

#define NATOM    100000
#define MNBR     12
#define NROWS    (NATOM*MNBR)
#define AF       64
#define NBRF     41
#define NBRF_PAD 48
#define ORIGF    92
#define EMBF     128
#define NCRYS    2048
#define CONV_K   192      // 64 + 64 + 48(padded nbr) -> 6 k-steps of 32
#define TOT_K    169      // 2*AF + NBRF
#define A_STRIDE 200      // halves; padded vs 192 for LDS bank spread
#define E_STRIDE 104      // halves; padded vs 96
#define BN_EPS   1e-5f

// ---------------- device helpers ----------------

__device__ inline float softplusf(float x) {
  return fmaxf(x, 0.f) + log1pf(__expf(-fabsf(x)));
}
__device__ inline float sigmoidf(float x) { return 1.f / (1.f + __expf(-x)); }

// Build one 16x32 f16 fragment from an LDS row, per CDNA5 WMMA A/B layout:
// lanes 0-15 hold K = k0+0..7 and k0+16..23 ; lanes 16-31 hold K = k0+8..15 and k0+24..31.
__device__ inline v16h frag_ld(const _Float16* rowPtr, int k0, int hi) {
  const v8h* p = (const v8h*)(rowPtr + k0 + hi * 8);
  v8h lo = p[0];
  v8h up = p[2];   // +16 halves
  v16h r;
#pragma unroll
  for (int i = 0; i < 8; ++i) { r[i] = lo[i]; r[i + 8] = up[i]; }
  return r;
}

__device__ inline v8f wmma_f16(v16h a, v16h b, v8f c) {
  return __builtin_amdgcn_wmma_f32_16x16x32_f16(false, a, false, b, (short)0, c,
                                                false, false);
}

// Stage pre-transposed, pre-padded weights Wt[col][CONV_K] f16 into LDS.
__device__ inline void stage_W(const _Float16* __restrict__ Wt, _Float16* sB, int tid) {
  for (int c = tid; c < 128 * (CONV_K / 8); c += 256) {
    int col = c / (CONV_K / 8), seg = c - col * (CONV_K / 8);
    *(v8h*)&sB[col * A_STRIDE + seg * 8] = *(const v8h*)&Wt[(size_t)col * CONV_K + seg * 8];
  }
}

// Asynchronously build a 64-row A tile in LDS:
// [self 64 | gathered-neighbor 64 | nbr_fea 48 | pad 16] as f16, via
// GLOBAL_LOAD_ASYNC_TO_LDS_B128 (ASYNCcnt). Exactly 6 issues per thread/wave.
__device__ inline void stage_A_async(const _Float16* __restrict__ atom_f16,
                                     const _Float16* __restrict__ nbr_f16,
                                     const _Float16* __restrict__ zbuf,
                                     const int* __restrict__ idx,
                                     _Float16* sA, int rowBase, int tid) {
#pragma unroll
  for (int it = 0; it < 6; ++it) {
    int c = tid + it * 256;            // c in [0, 64*24)
    int rl = c / 24, seg = c - rl * 24;
    int r = rowBase + rl;
    const _Float16* gp;
    if (seg < 8) {
      gp = &atom_f16[(size_t)(r / MNBR) * AF + seg * 8];
    } else if (seg < 16) {
      gp = &atom_f16[(size_t)idx[r] * AF + (seg - 8) * 8];
    } else if (seg < 22) {
      gp = &nbr_f16[(size_t)r * NBRF_PAD + (seg - 16) * 8];
    } else {
      gp = zbuf;                        // K-pad: stream zeros from L2-hot buffer
    }
    unsigned lds = (unsigned)(uintptr_t)(void*)&sA[rl * A_STRIDE + seg * 8];
    unsigned long long ga = (unsigned long long)(uintptr_t)gp;
    asm volatile("global_load_async_to_lds_b128 %0, %1, off"
                 :: "v"(lds), "v"(ga) : "memory");
  }
}

__device__ inline void wait_async_le6() {
  asm volatile("s_wait_asynccnt 0x6" ::: "memory");
}
__device__ inline void wait_async_le0() {
  asm volatile("s_wait_asynccnt 0x0" ::: "memory");
}

// 16x64 output per wave: 1 A frag + 4 B frags + 4 WMMA per k-step, 6 k-steps.
__device__ inline void conv_gemm(const _Float16* sA, const _Float16* sB,
                                 int rowTile, int colHalf, int lane, v8f acc[4]) {
  int hi = lane >> 4, lr = lane & 15;
  const _Float16* arow = sA + (rowTile * 16 + lr) * A_STRIDE;
  const _Float16* brow = sB + (colHalf * 64 + lr) * A_STRIDE;
#pragma unroll
  for (int kk = 0; kk < 6; ++kk) {
    int k0 = kk * 32;
    v16h a = frag_ld(arow, k0, hi);
#pragma unroll
    for (int t = 0; t < 4; ++t) {
      v16h b = frag_ld(brow + t * 16 * A_STRIDE, k0, hi);
      acc[t] = wmma_f16(a, b, acc[t]);
    }
  }
}

// ---------------- kernels ----------------

// nbr_fea fp32 (N*M x 41) -> f16 padded to 48
__global__ __launch_bounds__(256) void k_cvt_nbr(const float* __restrict__ nbr,
                                                 _Float16* __restrict__ out) {
  size_t total = (size_t)NROWS * NBRF_PAD;
  for (size_t i = (size_t)blockIdx.x * 256 + threadIdx.x; i < total;
       i += (size_t)gridDim.x * 256) {
    size_t r = i / NBRF_PAD;
    int j = (int)(i - r * NBRF_PAD);
    out[i] = (j < NBRF) ? (_Float16)nbr[r * NBRF + j] : (_Float16)0;
  }
}

// fc_w (3 x 169 x 128) fp32 -> Wt (3 x 128 x 192) f16, transposed + padded
__global__ __launch_bounds__(256) void k_prep_w(const float* __restrict__ fcw,
                                                _Float16* __restrict__ Wt) {
  size_t total = (size_t)3 * 128 * CONV_K;
  for (size_t i = (size_t)blockIdx.x * 256 + threadIdx.x; i < total;
       i += (size_t)gridDim.x * 256) {
    int layer = (int)(i / (128 * CONV_K));
    int rem = (int)(i - (size_t)layer * 128 * CONV_K);
    int f = rem / CONV_K, k = rem - f * CONV_K;
    const float* W = fcw + (size_t)layer * TOT_K * 128;
    Wt[i] = (k < TOT_K) ? (_Float16)W[(size_t)k * 128 + f] : (_Float16)0;
  }
}

__global__ __launch_bounds__(256) void k_zero(float* __restrict__ p, int n) {
  for (int i = blockIdx.x * 256 + threadIdx.x; i < n; i += gridDim.x * 256) p[i] = 0.f;
}

// atom_fea = x @ in_w + in_b  (WMMA), plus f16 shadow
__global__ __launch_bounds__(256) void k_embed(const float* __restrict__ x,
                                               const float* __restrict__ in_w,
                                               const float* __restrict__ in_b,
                                               float* __restrict__ atom_fea,
                                               _Float16* __restrict__ atom_f16) {
  __shared__ __align__(16) _Float16 sA[128 * E_STRIDE];
  __shared__ __align__(16) _Float16 sB[64 * E_STRIDE];
  int tid = threadIdx.x, lane = tid & 31, wave = tid >> 5;

  // stage in_w^T (f16, K padded to 96)
  for (int c = tid; c < 64 * 24; c += 256) {
    int nf = c / 24, seg = c - nf * 24, k = seg * 4;
    v4h o;
#pragma unroll
    for (int j = 0; j < 4; ++j) {
      int kk = k + j;
      o[j] = (kk < ORIGF) ? (_Float16)in_w[(size_t)kk * AF + nf] : (_Float16)0;
    }
    *(v4h*)&sB[nf * E_STRIDE + k] = o;
  }

  int numTiles = (NATOM + 127) / 128;
  for (int tile = blockIdx.x; tile < numTiles; tile += gridDim.x) {
    int rowBase = tile * 128;
    __syncthreads();
    for (int c = tid; c < 128 * 24; c += 256) {
      int rl = c / 24, seg = c - rl * 24, k = seg * 4;
      int rg = rowBase + rl;
      v4h o;
      if (rg < NATOM && seg < 23) {
        const float4 v = *(const float4*)&x[(size_t)rg * ORIGF + k];
        o[0] = (_Float16)v.x; o[1] = (_Float16)v.y;
        o[2] = (_Float16)v.z; o[3] = (_Float16)v.w;
      } else {
        o[0] = o[1] = o[2] = o[3] = (_Float16)0;
      }
      *(v4h*)&sA[rl * E_STRIDE + k] = o;
    }
    __syncthreads();

    int hi = lane >> 4, lr = lane & 15;
    v8f acc[4];
#pragma unroll
    for (int t = 0; t < 4; ++t)
#pragma unroll
      for (int r = 0; r < 8; ++r) acc[t][r] = 0.f;

    const _Float16* arow = &sA[(wave * 16 + lr) * E_STRIDE];
#pragma unroll
    for (int kk = 0; kk < 3; ++kk) {
      v16h a = frag_ld(arow, kk * 32, hi);
#pragma unroll
      for (int t = 0; t < 4; ++t) {
        v16h b = frag_ld(&sB[(t * 16 + lr) * E_STRIDE], kk * 32, hi);
        acc[t] = wmma_f16(a, b, acc[t]);
      }
    }
#pragma unroll
    for (int t = 0; t < 4; ++t) {
      int col = t * 16 + lr;
      float bia = in_b[col];
#pragma unroll
      for (int r = 0; r < 8; ++r) {
        int rg = rowBase + wave * 16 + hi * 8 + r;
        if (rg < NATOM) {
          float v = acc[t][r] + bia;
          atom_fea[(size_t)rg * AF + col] = v;
          atom_f16[(size_t)rg * AF + col] = (_Float16)v;
        }
      }
    }
  }
}

// Pass 1: GEMM (async-staged, double-buffered A), accumulate BN1 sum/sumsq.
__global__ __launch_bounds__(256) void k_conv_pass1(
    const _Float16* __restrict__ atom_f16, const _Float16* __restrict__ nbr_f16,
    const _Float16* __restrict__ zbuf, const int* __restrict__ idx,
    const _Float16* __restrict__ Wt, float* __restrict__ gstats) {
  __shared__ __align__(16) _Float16 sA[2][64 * A_STRIDE];
  __shared__ __align__(16) _Float16 sB[128 * A_STRIDE];
  __shared__ float spart[256];
  int tid = threadIdx.x, lane = tid & 31, wave = tid >> 5;
  int rowTile = wave & 3, colHalf = wave >> 2;
  stage_W(Wt, sB, tid);
  int numTiles = NROWS / 64;
  int buf = 0;
  int tile = blockIdx.x;
  if (tile < numTiles)
    stage_A_async(atom_f16, nbr_f16, zbuf, idx, sA[0], tile * 64, tid);
  for (; tile < numTiles; tile += gridDim.x, buf ^= 1) {
    int nxt = tile + gridDim.x;
    if (nxt < numTiles) {
      stage_A_async(atom_f16, nbr_f16, zbuf, idx, sA[buf ^ 1], nxt * 64, tid);
      wait_async_le6();   // current buffer complete; next tile in flight
    } else {
      wait_async_le0();
    }
    spart[tid] = 0.f;
    __syncthreads();

    v8f acc[4];
#pragma unroll
    for (int t = 0; t < 4; ++t)
#pragma unroll
      for (int r = 0; r < 8; ++r) acc[t][r] = 0.f;
    conv_gemm(sA[buf], sB, rowTile, colHalf, lane, acc);

#pragma unroll
    for (int t = 0; t < 4; ++t) {
      int col = colHalf * 64 + t * 16 + (lane & 15);
      float s = 0.f, q = 0.f;
#pragma unroll
      for (int r = 0; r < 8; ++r) { float y = acc[t][r]; s += y; q += y * y; }
      atomicAdd(&spart[col], s);        // ds_add_f32
      atomicAdd(&spart[128 + col], q);  // ds_add_f32
    }
    __syncthreads();
    atomicAdd(&gstats[tid], spart[tid]);
  }
}

// fold BN1: bias shifts mean only (variance is shift-invariant)
__global__ void k_fold1(const float* __restrict__ gstats, const float* __restrict__ fcb,
                        const float* __restrict__ g, const float* __restrict__ b,
                        float* __restrict__ scale, float* __restrict__ shift) {
  int t = threadIdx.x;
  if (t < 128) {
    const float inv_cnt = 1.0f / (float)NROWS;
    float m0 = gstats[t] * inv_cnt;
    float var = gstats[128 + t] * inv_cnt - m0 * m0;
    float mean = m0 + fcb[t];
    float sc = g[t] * rsqrtf(var + BN_EPS);
    scale[t] = sc;
    shift[t] = b[t] - mean * sc;
  }
}

// Pass 2: recompute GEMM (async double-buffered), BN1, gate, scatter-sum over M.
__global__ __launch_bounds__(256) void k_conv_pass2(
    const _Float16* __restrict__ atom_f16, const _Float16* __restrict__ nbr_f16,
    const _Float16* __restrict__ zbuf, const int* __restrict__ idx,
    const _Float16* __restrict__ Wt, const float* __restrict__ scale1,
    const float* __restrict__ shift1, float* __restrict__ nbr_sum) {
  __shared__ __align__(16) _Float16 sA[2][64 * A_STRIDE];
  __shared__ __align__(16) _Float16 sB[128 * A_STRIDE];
  __shared__ float sY[64 * 132];
  int tid = threadIdx.x, lane = tid & 31, wave = tid >> 5;
  int rowTile = wave & 3, colHalf = wave >> 2;
  stage_W(Wt, sB, tid);
  int numTiles = NROWS / 64;
  int buf = 0;
  int tile = blockIdx.x;
  if (tile < numTiles)
    stage_A_async(atom_f16, nbr_f16, zbuf, idx, sA[0], tile * 64, tid);
  for (; tile < numTiles; tile += gridDim.x, buf ^= 1) {
    int rowBase = tile * 64;
    int nxt = tile + gridDim.x;
    if (nxt < numTiles) {
      stage_A_async(atom_f16, nbr_f16, zbuf, idx, sA[buf ^ 1], nxt * 64, tid);
      wait_async_le6();
    } else {
      wait_async_le0();
    }
    __syncthreads();

    v8f acc[4];
#pragma unroll
    for (int t = 0; t < 4; ++t)
#pragma unroll
      for (int r = 0; r < 8; ++r) acc[t][r] = 0.f;
    conv_gemm(sA[buf], sB, rowTile, colHalf, lane, acc);

    int hi = lane >> 4;
#pragma unroll
    for (int t = 0; t < 4; ++t) {
      int col = colHalf * 64 + t * 16 + (lane & 15);
      float sc = scale1[col], sh = shift1[col];
#pragma unroll
      for (int r = 0; r < 8; ++r)
        sY[(rowTile * 16 + hi * 8 + r) * 132 + col] = acc[t][r] * sc + sh;
    }
    __syncthreads();
    for (int c = tid; c < 64 * 64; c += 256) {
      int rl = c >> 6, f = c & 63;
      float flt = sY[rl * 132 + f];
      float cor = sY[rl * 132 + 64 + f];
      float gv = sigmoidf(flt) * softplusf(cor);
      int n = (rowBase + rl) / MNBR;
      atomicAdd(&nbr_sum[(size_t)n * AF + f], gv);
    }
  }
}

// BN2 stats over atoms (stride of total threads is a multiple of 64 -> fixed f per thread)
__global__ __launch_bounds__(256) void k_stats2(const float* __restrict__ nbr_sum,
                                                float* __restrict__ g2) {
  size_t gt = (size_t)blockIdx.x * 256 + threadIdx.x;
  size_t stride = (size_t)gridDim.x * 256;
  int f = (int)(gt & 63);
  float s = 0.f, q = 0.f;
  for (size_t i = gt; i < (size_t)NATOM * AF; i += stride) {
    float v = nbr_sum[i];
    s += v;
    q += v * v;
  }
  atomicAdd(&g2[f], s);
  atomicAdd(&g2[64 + f], q);
}

__global__ void k_fold2(const float* __restrict__ g2, const float* __restrict__ g,
                        const float* __restrict__ b, float* __restrict__ scale,
                        float* __restrict__ shift) {
  int t = threadIdx.x;
  if (t < 64) {
    const float inv_cnt = 1.0f / (float)NATOM;
    float m0 = g2[t] * inv_cnt;
    float var = g2[64 + t] * inv_cnt - m0 * m0;
    float sc = g[t] * rsqrtf(var + BN_EPS);
    scale[t] = sc;
    shift[t] = b[t] - m0 * sc;
  }
}

// atom_fea = softplus(atom_fea + bn2(nbr_sum)); refresh f16 shadow
__global__ __launch_bounds__(256) void k_update(const float* __restrict__ nbr_sum,
                                                const float* __restrict__ scale2,
                                                const float* __restrict__ shift2,
                                                float* __restrict__ atom_fea,
                                                _Float16* __restrict__ atom_f16) {
  for (size_t i = (size_t)blockIdx.x * 256 + threadIdx.x; i < (size_t)NATOM * AF;
       i += (size_t)gridDim.x * 256) {
    int f = (int)(i & 63);
    float ns = nbr_sum[i] * scale2[f] + shift2[f];
    float v = softplusf(atom_fea[i] + ns);
    atom_fea[i] = v;
    atom_f16[i] = (_Float16)v;
  }
}

__global__ __launch_bounds__(256) void k_seg(const float* __restrict__ atom_fea,
                                             const int* __restrict__ batch,
                                             float* __restrict__ crys_sum,
                                             float* __restrict__ crys_cnt) {
  for (size_t i = (size_t)blockIdx.x * 256 + threadIdx.x; i < (size_t)NATOM * AF;
       i += (size_t)gridDim.x * 256) {
    int n = (int)(i >> 6), f = (int)(i & 63);
    int c = batch[n];
    atomicAdd(&crys_sum[(size_t)c * AF + f], atom_fea[i]);
    if (f == 0) atomicAdd(&crys_cnt[c], 1.0f);
  }
}

// per-crystal: mean -> softplus -> @cf_w+cf_b -> softplus -> @out_w+out_b
__global__ __launch_bounds__(128) void k_head(const float* __restrict__ crys_sum,
                                              const float* __restrict__ crys_cnt,
                                              const float* __restrict__ cf_w,
                                              const float* __restrict__ cf_b,
                                              const float* __restrict__ out_w,
                                              const float* __restrict__ out_b,
                                              float* __restrict__ out) {
  __shared__ float sc[64];
  __shared__ float red[128];
  int c = blockIdx.x, t = threadIdx.x;
  if (t < 64) {
    float cnt = fmaxf(crys_cnt[c], 1.0f);
    sc[t] = softplusf(crys_sum[(size_t)c * AF + t] / cnt);
  }
  __syncthreads();
  float acc = cf_b[t];
  for (int k = 0; k < 64; ++k) acc += sc[k] * cf_w[(size_t)k * EMBF + t];
  red[t] = softplusf(acc) * out_w[t];
  __syncthreads();
  for (int s = 64; s > 0; s >>= 1) {
    if (t < s) red[t] += red[t + s];
    __syncthreads();
  }
  if (t == 0) out[c] = red[0] + out_b[0];
}

// ---------------- host launch ----------------

extern "C" void kernel_launch(void* const* d_in, const int* in_sizes, int n_in,
                              void* d_out, int out_size, void* d_ws, size_t ws_size,
                              hipStream_t stream) {
  (void)in_sizes; (void)n_in; (void)out_size; (void)ws_size;
  const float* x      = (const float*)d_in[0];
  const float* nbrF   = (const float*)d_in[1];
  const int*   idx    = (const int*)d_in[2];
  const int*   batch  = (const int*)d_in[3];
  const float* in_w   = (const float*)d_in[4];
  const float* in_b   = (const float*)d_in[5];
  const float* fc_w   = (const float*)d_in[6];
  const float* fc_b   = (const float*)d_in[7];
  const float* bn1_g  = (const float*)d_in[8];
  const float* bn1_b  = (const float*)d_in[9];
  const float* bn2_g  = (const float*)d_in[10];
  const float* bn2_b  = (const float*)d_in[11];
  const float* cf_w   = (const float*)d_in[12];
  const float* cf_b   = (const float*)d_in[13];
  const float* out_w  = (const float*)d_in[14];
  const float* out_b  = (const float*)d_in[15];
  float* out = (float*)d_out;

  char* ws = (char*)d_ws;
  size_t off = 0;
  auto take = [&](size_t bytes) -> char* {
    char* p = ws + off;
    off = (off + bytes + 255) & ~(size_t)255;
    return p;
  };
  float*     atom_fea = (float*)take((size_t)NATOM * AF * 4);
  _Float16*  atom_f16 = (_Float16*)take((size_t)NATOM * AF * 2);
  _Float16*  nbr_f16  = (_Float16*)take((size_t)NROWS * NBRF_PAD * 2);
  _Float16*  Wt       = (_Float16*)take((size_t)3 * 128 * CONV_K * 2);
  float*     nbr_sum  = (float*)take((size_t)NATOM * AF * 4);
  float*     gstats   = (float*)take(256 * 4);
  float*     scale1   = (float*)take(128 * 4);
  float*     shift1   = (float*)take(128 * 4);
  float*     g2       = (float*)take(128 * 4);
  float*     scale2   = (float*)take(64 * 4);
  float*     shift2   = (float*)take(64 * 4);
  float*     crys_sum = (float*)take((size_t)NCRYS * AF * 4);
  float*     crys_cnt = (float*)take(NCRYS * 4);
  float*     zbuf     = (float*)take(64);          // 16-byte-aligned zeros for K-pad

  dim3 B(256);
  k_zero<<<1, B, 0, stream>>>(zbuf, 16);
  k_cvt_nbr<<<2048, B, 0, stream>>>(nbrF, nbr_f16);
  k_prep_w<<<288, B, 0, stream>>>(fc_w, Wt);
  k_embed<<<782, B, 0, stream>>>(x, in_w, in_b, atom_fea, atom_f16);

  const _Float16* zh = (const _Float16*)zbuf;
  for (int L = 0; L < 3; ++L) {
    const _Float16* WtL = Wt + (size_t)L * 128 * CONV_K;
    k_zero<<<1, B, 0, stream>>>(gstats, 256);
    k_zero<<<2048, B, 0, stream>>>(nbr_sum, NATOM * AF);
    k_zero<<<1, B, 0, stream>>>(g2, 128);
    k_conv_pass1<<<2048, B, 0, stream>>>(atom_f16, nbr_f16, zh, idx, WtL, gstats);
    k_fold1<<<1, 128, 0, stream>>>(gstats, fc_b + L * 128, bn1_g + L * 128,
                                   bn1_b + L * 128, scale1, shift1);
    k_conv_pass2<<<2048, B, 0, stream>>>(atom_f16, nbr_f16, zh, idx, WtL, scale1,
                                         shift1, nbr_sum);
    k_stats2<<<1024, B, 0, stream>>>(nbr_sum, g2);
    k_fold2<<<1, 64, 0, stream>>>(g2, bn2_g + L * 64, bn2_b + L * 64, scale2, shift2);
    k_update<<<2048, B, 0, stream>>>(nbr_sum, scale2, shift2, atom_fea, atom_f16);
  }

  k_zero<<<128, B, 0, stream>>>(crys_sum, NCRYS * AF);
  k_zero<<<8, B, 0, stream>>>(crys_cnt, NCRYS);
  k_seg<<<2048, B, 0, stream>>>(atom_fea, batch, crys_sum, crys_cnt);
  k_head<<<NCRYS, 128, 0, stream>>>(crys_sum, crys_cnt, cf_w, cf_b, out_w, out_b, out);
}
